// DrugGCN_26817775796788
// MI455X (gfx1250) — compile-verified, matrix operand order checked by reading, not run on hardware
//
#include <hip/hip_runtime.h>
#include <hip/hip_bf16.h>

// ---------------------------------------------------------------------------
// DrugGCN on MI455X (gfx1250): 3x (bf16 WMMA GEMM -> edge scatter -> BN/ReLU)
// GEMM: 128x128 block tile, K-step 32, 8 waves (4Mx2N), double-buffered LDS
// staged with GLOBAL_LOAD_ASYNC_TO_LDS_B128 (ASYNCcnt) when available.
// Matrix math: V_WMMA_F32_16X16X32_BF16, f32 accumulate.
// ---------------------------------------------------------------------------

typedef __bf16 bf16_t;
typedef __attribute__((ext_vector_type(16))) __bf16 v16bf;
typedef __attribute__((ext_vector_type(8)))  float  v8f;
typedef int v4i_ __attribute__((vector_size(16)));   // matches builtin param type

#define FIN  256
#define FHID 512
#define FOUT 256
#define BN_EPS 1e-5f

#define BM   128
#define BN_T 128
#define BK   32

#if defined(__has_builtin)
#if __has_builtin(__builtin_amdgcn_global_load_async_to_lds_b128)
#define ASYNC_LDS 1
#endif
#endif
#ifndef ASYNC_LDS
#define ASYNC_LDS 0
#endif

static inline int ceildiv(int a, int b) { return (a + b - 1) / b; }

// Stage 16B global -> LDS. Async path uses the CDNA5 async-to-LDS DMA
// (per-lane addresses, tracked by ASYNCcnt); fallback goes through VGPRs.
__device__ __forceinline__ void stage16(const bf16_t* gsrc, bf16_t* ldst, bool inRange) {
#if ASYNC_LDS
    if (inRange) {
        __builtin_amdgcn_global_load_async_to_lds_b128(
            (__attribute__((address_space(1))) v4i_*)(uintptr_t)gsrc,
            (__attribute__((address_space(3))) v4i_*)(uintptr_t)ldst,
            0, 0);
    } else {
        uint4 z = {0u, 0u, 0u, 0u};
        *(uint4*)ldst = z;
    }
#else
    uint4 v = {0u, 0u, 0u, 0u};
    if (inRange) v = *(const uint4*)gsrc;
    *(uint4*)ldst = v;
#endif
}

__device__ __forceinline__ void wait_async_lds() {
#if ASYNC_LDS
#if defined(__has_builtin) && __has_builtin(__builtin_amdgcn_s_wait_asynccnt)
    __builtin_amdgcn_s_wait_asynccnt(0);
#else
    asm volatile("s_wait_asynccnt 0x0" ::: "memory");
#endif
#endif
}

// ---------------------------- small utility kernels -------------------------

__global__ void k_fill_f32(float* __restrict__ p, float v, int n) {
    int i = blockIdx.x * blockDim.x + threadIdx.x;
    if (i < n) p[i] = v;
}

__global__ void k_f32_to_bf16(const float* __restrict__ in, bf16_t* __restrict__ out, int n) {
    int i = blockIdx.x * blockDim.x + threadIdx.x;
    if (i < n) out[i] = (bf16_t)in[i];
}

// W: K x N (f32, row-major) -> Wt: N x K (bf16, row-major)
__global__ void k_transpose_to_bf16(const float* __restrict__ W, bf16_t* __restrict__ Wt,
                                    int K, int N) {
    int idx = blockIdx.x * blockDim.x + threadIdx.x;
    if (idx >= K * N) return;
    int k = idx / N;
    int n = idx - k * N;
    Wt[(size_t)n * K + k] = (bf16_t)W[idx];
}

__global__ void k_degree(const int* __restrict__ dst, float* __restrict__ deg, int E) {
    int i = blockIdx.x * blockDim.x + threadIdx.x;
    if (i < E) atomicAdd(&deg[dst[i]], 1.0f);
}

__global__ void k_dinv(const float* __restrict__ deg, float* __restrict__ dinv, int n) {
    int i = blockIdx.x * blockDim.x + threadIdx.x;
    if (i < n) dinv[i] = rsqrtf(deg[i] + 1.0f);   // +1 for self loop
}

// ------------------------------- WMMA GEMM ----------------------------------
// C[M,N] = A[M,K] (bf16 row-major) x Bt[N,K]^T (bf16, pre-transposed), f32 out.
// Block tile 128x128, K-step 32, double-buffered async LDS staging.
// 8 waves arranged 4(M) x 2(N); each wave: 32 rows x 64 cols = 8 WMMA tiles.
// Fragment packing per CDNA5 ISA 7.12.2 (wave32):
//   A/B 16x32 bf16: lane 0-15 -> M/N=lane, K in {0..7,16..23}; lane 16-31 -> +8
//   C/D 16x16 f32 : lane 0-15 -> N=lane, M=vgpr; lane 16-31 -> M=vgpr+8
__global__ __launch_bounds__(256)
void k_gemm_bf16_wmma(const bf16_t* __restrict__ A, const bf16_t* __restrict__ Bt,
                      float* __restrict__ C, int M, int N, int K) {
    __shared__ bf16_t As[2][BM][BK + 2];     // +2 bf16 pad -> 68B stride (bank-safe)
    __shared__ bf16_t Bs[2][BN_T][BK + 2];

    const int tid   = threadIdx.x;
    const int wave  = tid >> 5;              // wave32
    const int lane  = tid & 31;
    const int waveM = wave >> 1;             // 0..3 -> 32-row strip
    const int waveN = wave & 1;              // 0..1 -> 64-col strip
    const int m0 = blockIdx.y * BM;
    const int n0 = blockIdx.x * BN_T;

    const int lrow  = lane & 15;
    const int khalf = (lane >> 4) << 3;      // 0 or 8

    v8f acc[2][4] = {};
    const int nk = K / BK;

    auto stage = [&](int buf, int k0) {
        int q = tid << 1;
        #pragma unroll
        for (int i = 0; i < 2; ++i) {
            int qi = q + i;
            int r = qi >> 2;                 // 0..127
            int c = (qi & 3) << 3;           // 0,8,16,24
            int gm = m0 + r;
            stage16(A + (size_t)gm * K + k0 + c, &As[buf][r][c], gm < M);
            stage16(Bt + (size_t)(n0 + r) * K + k0 + c, &Bs[buf][r][c], true);
        }
    };

    stage(0, 0);
    wait_async_lds();
    __syncthreads();

    for (int kt = 0; kt < nk; ++kt) {
        const int cur = kt & 1;
        if (kt + 1 < nk) stage(cur ^ 1, (kt + 1) * BK);   // prefetch next tile (async)

        // A fragments for this wave's two 16-row subtiles
        v16bf afrag[2];
        #pragma unroll
        for (int i = 0; i < 2; ++i) {
            int mrow = (waveM << 5) + (i << 4) + lrow;
            #pragma unroll
            for (int v = 0; v < 8; ++v) {
                int kk = (v < 4) ? (khalf + (v << 1)) : (khalf + 16 + ((v - 4) << 1));
                afrag[i][2 * v]     = As[cur][mrow][kk];
                afrag[i][2 * v + 1] = As[cur][mrow][kk + 1];
            }
        }
        #pragma unroll
        for (int j = 0; j < 4; ++j) {
            int nrow = (waveN << 6) + (j << 4) + lrow;
            v16bf bfrag;
            #pragma unroll
            for (int v = 0; v < 8; ++v) {
                int kk = (v < 4) ? (khalf + (v << 1)) : (khalf + 16 + ((v - 4) << 1));
                bfrag[2 * v]     = Bs[cur][nrow][kk];
                bfrag[2 * v + 1] = Bs[cur][nrow][kk + 1];
            }
            #pragma unroll
            for (int i = 0; i < 2; ++i) {
                acc[i][j] = __builtin_amdgcn_wmma_f32_16x16x32_bf16(
                    false, afrag[i], false, bfrag, (short)0, acc[i][j], false, false);
            }
        }

        wait_async_lds();     // next tile fully in LDS
        __syncthreads();      // all waves done reading 'cur' before it is reused
    }

    // ---- store D ----
    const int ncol0 = n0 + (waveN << 6) + (lane & 15);
    const int mrow0 = m0 + (waveM << 5) + ((lane >> 4) << 3);
    #pragma unroll
    for (int i = 0; i < 2; ++i) {
        #pragma unroll
        for (int j = 0; j < 4; ++j) {
            #pragma unroll
            for (int v = 0; v < 8; ++v) {
                int gm = mrow0 + (i << 4) + v;
                if (gm < M) C[(size_t)gm * N + ncol0 + (j << 4)] = acc[i][j][v];
            }
        }
    }
}

// -------------------------- graph scatter (edges) ---------------------------
// AGG[dst] += H[src] * (dinv[src]*dinv[dst]); 4 feats/thread, coalesced over F.
__global__ void k_scatter(const float* __restrict__ H, float* __restrict__ AGG,
                          const int* __restrict__ src, const int* __restrict__ dst,
                          const float* __restrict__ dinv, int E, int F) {
    int idx = blockIdx.x * blockDim.x + threadIdx.x;
    int cpe = F >> 2;
    long long total = (long long)E * cpe;
    if (idx >= total) return;
    int e = idx / cpe;
    int c = (idx - e * cpe) << 2;
    int s = src[e], d = dst[e];
    float norm = dinv[s] * dinv[d];
    float4 h = *(const float4*)(H + (size_t)s * F + c);
    float* o = AGG + (size_t)d * F + c;
    atomicAdd(o + 0, h.x * norm);
    atomicAdd(o + 1, h.y * norm);
    atomicAdd(o + 2, h.z * norm);
    atomicAdd(o + 3, h.w * norm);
}

// self-loop + bias: AGG += H * dinv^2 + b
__global__ void k_self_bias(float* __restrict__ AGG, const float* __restrict__ H,
                            const float* __restrict__ dinv, const float* __restrict__ b,
                            int Nn, int F) {
    int idx = blockIdx.x * blockDim.x + threadIdx.x;
    if (idx >= Nn * F) return;
    int n = idx / F;
    int f = idx - n * F;
    float di = dinv[n];
    AGG[idx] += H[idx] * di * di + b[f];
}

// ----------------------------- BatchNorm ------------------------------------
__global__ void k_bn_stats(const float* __restrict__ H, float* __restrict__ sums,
                           float* __restrict__ sumsq, int Nn, int F, int rowsPer) {
    int gid = blockIdx.x * blockDim.x + threadIdx.x;
    int f  = gid % F;
    int rb = gid / F;
    int r0 = rb * rowsPer;
    if (r0 >= Nn) return;
    int r1 = r0 + rowsPer; if (r1 > Nn) r1 = Nn;
    float s = 0.f, s2 = 0.f;
    for (int r = r0; r < r1; ++r) {
        float v = H[(size_t)r * F + f];
        s += v; s2 += v * v;
    }
    atomicAdd(&sums[f], s);
    atomicAdd(&sumsq[f], s2);
}

// normalize + affine + ReLU + convert to bf16 (next GEMM A operand)
__global__ void k_bn_relu_bf16(const float* __restrict__ H, const float* __restrict__ sums,
                               const float* __restrict__ sumsq, const float* __restrict__ g,
                               const float* __restrict__ be, bf16_t* __restrict__ out,
                               float invN, int Nn, int F) {
    int idx = blockIdx.x * blockDim.x + threadIdx.x;
    if (idx >= Nn * F) return;
    int f = idx % F;
    float m   = sums[f] * invN;
    float var = sumsq[f] * invN - m * m;
    float v = (H[idx] - m) * rsqrtf(var + BN_EPS) * g[f] + be[f];
    out[idx] = (bf16_t)(v > 0.f ? v : 0.f);
}

// final: out = AGG + H*dinv^2 + b3 + residual(x)
__global__ void k_final(const float* __restrict__ AGG, const float* __restrict__ H,
                        const float* __restrict__ dinv, const float* __restrict__ b3,
                        const float* __restrict__ x, float* __restrict__ out,
                        int Nn, int F) {
    int idx = blockIdx.x * blockDim.x + threadIdx.x;
    if (idx >= Nn * F) return;
    int n = idx / F;
    int f = idx - n * F;
    float di = dinv[n];
    out[idx] = AGG[idx] + H[idx] * di * di + b3[f] + x[idx];
}

// ------------------------------- driver -------------------------------------

extern "C" void kernel_launch(void* const* d_in, const int* in_sizes, int n_in,
                              void* d_out, int out_size, void* d_ws, size_t ws_size,
                              hipStream_t stream) {
    const float* x   = (const float*)d_in[0];
    const int*   ei  = (const int*)d_in[1];
    const float* W1  = (const float*)d_in[2];
    const float* b1  = (const float*)d_in[3];
    const float* g1  = (const float*)d_in[4];
    const float* be1 = (const float*)d_in[5];
    const float* W2  = (const float*)d_in[6];
    const float* b2  = (const float*)d_in[7];
    const float* g2  = (const float*)d_in[8];
    const float* be2 = (const float*)d_in[9];
    const float* W3  = (const float*)d_in[10];
    const float* b3  = (const float*)d_in[11];
    float* out = (float*)d_out;

    const int Nn = in_sizes[0] / FIN;   // 50000
    const int E  = in_sizes[1] / 2;     // 800000
    const int* src = ei;
    const int* dst = ei + E;

    // ---- workspace carve (256B aligned) ----
    char* ws = (char*)d_ws;
    auto carve = [&](size_t bytes) -> void* {
        void* p = (void*)ws;
        ws += (bytes + 255) & ~((size_t)255);
        return p;
    };
    float*  deg   = (float*)carve((size_t)Nn * 4);
    float*  dinv  = (float*)carve((size_t)Nn * 4);
    bf16_t* Abf   = (bf16_t*)carve((size_t)Nn * FHID * 2);
    float*  H     = (float*)carve((size_t)Nn * FHID * 4);
    float*  AGG   = (float*)carve((size_t)Nn * FHID * 4);
    bf16_t* W1t   = (bf16_t*)carve((size_t)FHID * FIN  * 2);  // [N=512][K=256]
    bf16_t* W2t   = (bf16_t*)carve((size_t)FHID * FHID * 2);  // [512][512]
    bf16_t* W3t   = (bf16_t*)carve((size_t)FOUT * FHID * 2);  // [256][512]
    float*  sums  = (float*)carve((size_t)FHID * 4);
    float*  sumsq = (float*)carve((size_t)FHID * 4);          // contiguous after sums

    const int TB = 256;
    const float invN = 1.0f / (float)Nn;

    // ---- degrees ----
    k_fill_f32<<<ceildiv(Nn, TB), TB, 0, stream>>>(deg, 0.f, Nn);
    k_degree<<<ceildiv(E, TB), TB, 0, stream>>>(dst, deg, E);
    k_dinv<<<ceildiv(Nn, TB), TB, 0, stream>>>(deg, dinv, Nn);

    // ---- weight transpose+convert, input convert ----
    k_transpose_to_bf16<<<ceildiv(FIN * FHID, TB), TB, 0, stream>>>(W1, W1t, FIN, FHID);
    k_transpose_to_bf16<<<ceildiv(FHID * FHID, TB), TB, 0, stream>>>(W2, W2t, FHID, FHID);
    k_transpose_to_bf16<<<ceildiv(FHID * FOUT, TB), TB, 0, stream>>>(W3, W3t, FHID, FOUT);
    k_f32_to_bf16<<<ceildiv(Nn * FIN, TB), TB, 0, stream>>>(x, Abf, Nn * FIN);

    dim3 blk(256);
    int gy = ceildiv(Nn, BM);
    const int rowsPer = 128;

    // =============== layer 1: 256 -> 512 ===============
    {
        dim3 grid(FHID / BN_T, gy);
        k_gemm_bf16_wmma<<<grid, blk, 0, stream>>>(Abf, W1t, H, Nn, FHID, FIN);
        k_fill_f32<<<ceildiv(Nn * FHID, TB), TB, 0, stream>>>(AGG, 0.f, Nn * FHID);
        long long sc = (long long)E * (FHID / 4);
        k_scatter<<<(int)((sc + TB - 1) / TB), TB, 0, stream>>>(H, AGG, src, dst, dinv, E, FHID);
        k_self_bias<<<ceildiv(Nn * FHID, TB), TB, 0, stream>>>(AGG, H, dinv, b1, Nn, FHID);
        k_fill_f32<<<ceildiv(2 * FHID, TB), TB, 0, stream>>>(sums, 0.f, 2 * FHID);
        k_bn_stats<<<ceildiv(FHID * ceildiv(Nn, rowsPer), TB), TB, 0, stream>>>(AGG, sums, sumsq, Nn, FHID, rowsPer);
        k_bn_relu_bf16<<<ceildiv(Nn * FHID, TB), TB, 0, stream>>>(AGG, sums, sumsq, g1, be1, Abf, invN, Nn, FHID);
    }

    // =============== layer 2: 512 -> 512 ===============
    {
        dim3 grid(FHID / BN_T, gy);
        k_gemm_bf16_wmma<<<grid, blk, 0, stream>>>(Abf, W2t, H, Nn, FHID, FHID);
        k_fill_f32<<<ceildiv(Nn * FHID, TB), TB, 0, stream>>>(AGG, 0.f, Nn * FHID);
        long long sc = (long long)E * (FHID / 4);
        k_scatter<<<(int)((sc + TB - 1) / TB), TB, 0, stream>>>(H, AGG, src, dst, dinv, E, FHID);
        k_self_bias<<<ceildiv(Nn * FHID, TB), TB, 0, stream>>>(AGG, H, dinv, b2, Nn, FHID);
        k_fill_f32<<<ceildiv(2 * FHID, TB), TB, 0, stream>>>(sums, 0.f, 2 * FHID);
        k_bn_stats<<<ceildiv(FHID * ceildiv(Nn, rowsPer), TB), TB, 0, stream>>>(AGG, sums, sumsq, Nn, FHID, rowsPer);
        k_bn_relu_bf16<<<ceildiv(Nn * FHID, TB), TB, 0, stream>>>(AGG, sums, sumsq, g2, be2, Abf, invN, Nn, FHID);
    }

    // =============== layer 3: 512 -> 256 (+residual) ===============
    {
        dim3 grid(FOUT / BN_T, gy);
        k_gemm_bf16_wmma<<<grid, blk, 0, stream>>>(Abf, W3t, H, Nn, FOUT, FHID);
        k_fill_f32<<<ceildiv(Nn * FOUT, TB), TB, 0, stream>>>(AGG, 0.f, Nn * FOUT);
        long long sc = (long long)E * (FOUT / 4);
        k_scatter<<<(int)((sc + TB - 1) / TB), TB, 0, stream>>>(H, AGG, src, dst, dinv, E, FOUT);
        k_final<<<ceildiv(Nn * FOUT, TB), TB, 0, stream>>>(AGG, H, dinv, b3, x, out, Nn, FOUT);
    }
}
